// EGNNLayer_11630771437665
// MI455X (gfx1250) — matrix operand order repaired and checked by reading, not compiled, not run on hardware
//
#include <hip/hip_runtime.h>

#define NN 10000
#define EE 320000
#define HH 256
#define KP1 544   // 2H+1 = 513 padded up to a multiple of 32
#define KU  512   // 2H for the update MLP
#define MT  64    // edges per workgroup in the edge kernel (4 x 16-row tiles)

typedef __attribute__((ext_vector_type(16))) __bf16 v16bf;
typedef __attribute__((ext_vector_type(8)))  float  v8f;

struct U4x2 { uint4 a, b; };

static __device__ __forceinline__ v16bf pack16(uint4 a, uint4 b) {
  U4x2 t{a, b};
  return __builtin_bit_cast(v16bf, t);
}

// fp32 -> bf16 round-to-nearest-even (bit trick; NaN edge ignored)
static __device__ __forceinline__ unsigned short f2bf(float f) {
  unsigned u = __builtin_bit_cast(unsigned, f);
  u += 0x7FFFu + ((u >> 16) & 1u);
  return (unsigned short)(u >> 16);
}

static __device__ __forceinline__ float silu_f(float x) {
  return x / (1.0f + __expf(-x));
}

static __device__ __forceinline__ v8f wmma_bf16(v16bf a, v16bf b, v8f c) {
  return __builtin_amdgcn_wmma_f32_16x16x32_bf16(false, a, false, b,
                                                 (short)0, c, false, false);
}

// ---------------------------------------------------------------------------
// Prep: x -> bf16 copy, zero aggr
// ---------------------------------------------------------------------------
__global__ void prep_x_kernel(const float* __restrict__ x,
                              unsigned short* __restrict__ xb,
                              float* __restrict__ aggr, int n) {
  int i = blockIdx.x * blockDim.x + threadIdx.x;
  if (i < n) {
    xb[i] = f2bf(x[i]);
    aggr[i] = 0.0f;
  }
}

// Prep: W [K x Nm] fp32 -> WT [Nm x Kp] bf16, zero-padded rows k >= K
__global__ void prep_wt_kernel(const float* __restrict__ src,
                               unsigned short* __restrict__ dst,
                               int K, int Nm, int Kp) {
  int i = blockIdx.x * blockDim.x + threadIdx.x;
  if (i < Nm * Kp) {
    int n = i / Kp, k = i - n * Kp;
    dst[i] = (k < K) ? f2bf(src[(size_t)k * Nm + n]) : (unsigned short)0;
  }
}

// ---------------------------------------------------------------------------
// Edge kernel: fused gather -> MLP(2 GEMMs via WMMA, SiLU) -> atomic scatter
// One 256-thread workgroup (8 waves) per 64-edge tile.
// Wave w owns 32 output columns; each B fragment is reused across the four
// 16-row A tiles (4x less weight traffic than a 16-edge tile).
// ---------------------------------------------------------------------------
__global__ void __launch_bounds__(256)
egnn_edge_kernel(const unsigned short* __restrict__ xb,
                 const float* __restrict__ pos,
                 const long long* __restrict__ ei,
                 const unsigned short* __restrict__ w1t,  // [256][KP1] bf16
                 const float* __restrict__ bm1,
                 const unsigned short* __restrict__ w2t,  // [256][256] bf16
                 const float* __restrict__ bm2,
                 float* __restrict__ aggr)
{
  __shared__ unsigned short s_state[MT * KP1];   // 69.6 KB
  __shared__ unsigned short s_hid[MT * HH];      // 32 KB
  __shared__ int s_rec[MT];

  const int t  = threadIdx.x;
  const int e0 = blockIdx.x * MT;

  // ---- gather: 4 threads per edge row, 128 bf16 each ----
  {
    const int row = t >> 2;          // 0..63
    const int sub = t & 3;           // 0..3
    const long long snd = ei[e0 + row];
    const long long rcv = ei[EE + e0 + row];
    const long long node = (sub < 2) ? snd : rcv;
    const unsigned short* src = xb + (size_t)node * HH + (size_t)(sub & 1) * 128;
    unsigned short* dst = s_state + row * KP1 + sub * 128;
#pragma unroll
    for (int i = 0; i < 16; ++i)
      ((uint4*)dst)[i] = ((const uint4*)src)[i];
  }
  if (t < MT) {
    const long long snd = ei[e0 + t];
    const long long rcv = ei[EE + e0 + t];
    s_rec[t] = (int)rcv;
    float dx = pos[snd * 3 + 0] - pos[rcv * 3 + 0];
    float dy = pos[snd * 3 + 1] - pos[rcv * 3 + 1];
    float dz = pos[snd * 3 + 2] - pos[rcv * 3 + 2];
    float dist = sqrtf(dx * dx + dy * dy + dz * dz);
    s_state[t * KP1 + 512] = f2bf(dist);
    for (int k = 513; k < KP1; ++k) s_state[t * KP1 + k] = 0;
  }
  __syncthreads();

  const int wave = t >> 5;
  const int lane = t & 31;
  const int hi   = lane >> 4;     // lane half selects K sub-stripe
  const int lo   = lane & 15;     // row (A) / col-in-tile (B,C)
  const int nb   = wave * 32;     // this wave's output-column base
  const int kbA  = hi * 8;
  const int kbB  = hi * 8;

  // ---- GEMM1: state[64 x 544] x Wm1[544 x 256] ----
  v8f acc[4][2] = {};
#pragma unroll
  for (int k0 = 0; k0 < KP1; k0 += 32) {
    const unsigned short* bp0 = w1t + (size_t)(nb + lo) * KP1 + k0 + kbB;
    const unsigned short* bp1 = w1t + (size_t)(nb + 16 + lo) * KP1 + k0 + kbB;
    v16bf b0 = pack16(*(const uint4*)bp0, *(const uint4*)(bp0 + 16));
    v16bf b1 = pack16(*(const uint4*)bp1, *(const uint4*)(bp1 + 16));
#pragma unroll
    for (int mt = 0; mt < 4; ++mt) {
      const unsigned short* ap = s_state + (mt * 16 + lo) * KP1 + k0 + kbA;
      v16bf a = pack16(*(const uint4*)ap, *(const uint4*)(ap + 16));
      acc[mt][0] = wmma_bf16(a, b0, acc[mt][0]);
      acc[mt][1] = wmma_bf16(a, b1, acc[mt][1]);
    }
  }
  {
    const float bia0 = bm1[nb + lo];
    const float bia1 = bm1[nb + 16 + lo];
#pragma unroll
    for (int mt = 0; mt < 4; ++mt)
#pragma unroll
      for (int r = 0; r < 8; ++r) {
        const int m = mt * 16 + r + hi * 8;
        s_hid[m * HH + nb + lo]      = f2bf(silu_f(acc[mt][0][r] + bia0));
        s_hid[m * HH + nb + 16 + lo] = f2bf(silu_f(acc[mt][1][r] + bia1));
      }
  }
  __syncthreads();

  // ---- GEMM2: hid[64 x 256] x Wm2[256 x 256] ----
#pragma unroll
  for (int mt = 0; mt < 4; ++mt) {
    acc[mt][0] = (v8f){};
    acc[mt][1] = (v8f){};
  }
#pragma unroll
  for (int k0 = 0; k0 < HH; k0 += 32) {
    const unsigned short* bp0 = w2t + (size_t)(nb + lo) * HH + k0 + kbB;
    const unsigned short* bp1 = w2t + (size_t)(nb + 16 + lo) * HH + k0 + kbB;
    v16bf b0 = pack16(*(const uint4*)bp0, *(const uint4*)(bp0 + 16));
    v16bf b1 = pack16(*(const uint4*)bp1, *(const uint4*)(bp1 + 16));
#pragma unroll
    for (int mt = 0; mt < 4; ++mt) {
      const unsigned short* ap = s_hid + (mt * 16 + lo) * HH + k0 + kbA;
      v16bf a = pack16(*(const uint4*)ap, *(const uint4*)(ap + 16));
      acc[mt][0] = wmma_bf16(a, b0, acc[mt][0]);
      acc[mt][1] = wmma_bf16(a, b1, acc[mt][1]);
    }
  }
  {
    const float bia0 = bm2[nb + lo];
    const float bia1 = bm2[nb + 16 + lo];
#pragma unroll
    for (int mt = 0; mt < 4; ++mt)
#pragma unroll
      for (int r = 0; r < 8; ++r) {
        const int m    = mt * 16 + r + hi * 8;
        const int dstn = s_rec[m];
        float m0 = silu_f(acc[mt][0][r] + bia0);
        float m1 = silu_f(acc[mt][1][r] + bia1);
        __hip_atomic_fetch_add(&aggr[(size_t)dstn * HH + nb + lo], m0,
                               __ATOMIC_RELAXED, __HIP_MEMORY_SCOPE_AGENT);
        __hip_atomic_fetch_add(&aggr[(size_t)dstn * HH + nb + 16 + lo], m1,
                               __ATOMIC_RELAXED, __HIP_MEMORY_SCOPE_AGENT);
      }
  }
}

// ---------------------------------------------------------------------------
// Node kernel: update = Wu2 * SiLU(Wu1 * [x|aggr] + bu1) + bu2  (M = 16)
// ---------------------------------------------------------------------------
__global__ void __launch_bounds__(256)
egnn_node_kernel(const unsigned short* __restrict__ xb,
                 const float* __restrict__ aggr,
                 const unsigned short* __restrict__ u1t,  // [256][512] bf16
                 const float* __restrict__ bu1,
                 const unsigned short* __restrict__ u2t,  // [256][256] bf16
                 const float* __restrict__ bu2,
                 float* __restrict__ out)
{
  __shared__ unsigned short s_state[16 * KU];
  __shared__ unsigned short s_hid[16 * HH];

  const int t  = threadIdx.x;
  const int n0 = blockIdx.x * 16;

  {
    const int row  = t >> 4;
    const int sub  = t & 15;
    const int node = n0 + row;
    unsigned short* dst = s_state + row * KU + sub * 32;
    if (sub < 8) {
      const unsigned short* src = xb + (size_t)node * HH + sub * 32;
#pragma unroll
      for (int i = 0; i < 4; ++i)
        ((uint4*)dst)[i] = ((const uint4*)src)[i];
    } else {
      const float* src = aggr + (size_t)node * HH + (sub - 8) * 32;
#pragma unroll
      for (int i = 0; i < 8; ++i) {
        float4 v = ((const float4*)src)[i];
        ushort4 o;
        o.x = f2bf(v.x); o.y = f2bf(v.y); o.z = f2bf(v.z); o.w = f2bf(v.w);
        ((ushort4*)dst)[i] = o;
      }
    }
  }
  __syncthreads();

  const int wave = t >> 5;
  const int lane = t & 31;
  const int hi   = lane >> 4;
  const int lo   = lane & 15;
  const int nb   = wave * 32;
  const int kbA  = hi * 8;
  const int kbB  = hi * 8;

  // ---- GEMM1: [16 x 512] x Wu1[512 x 256] ----
  v8f acc0 = {}, acc1 = {};
#pragma unroll
  for (int k0 = 0; k0 < KU; k0 += 32) {
    const unsigned short* ap = s_state + lo * KU + k0 + kbA;
    v16bf a = pack16(*(const uint4*)ap, *(const uint4*)(ap + 16));
    const unsigned short* bp0 = u1t + (size_t)(nb + lo) * KU + k0 + kbB;
    const unsigned short* bp1 = u1t + (size_t)(nb + 16 + lo) * KU + k0 + kbB;
    v16bf b0 = pack16(*(const uint4*)bp0, *(const uint4*)(bp0 + 16));
    v16bf b1 = pack16(*(const uint4*)bp1, *(const uint4*)(bp1 + 16));
    acc0 = wmma_bf16(a, b0, acc0);
    acc1 = wmma_bf16(a, b1, acc1);
  }
  {
    const float bia0 = bu1[nb + lo];
    const float bia1 = bu1[nb + 16 + lo];
#pragma unroll
    for (int r = 0; r < 8; ++r) {
      const int m = r + hi * 8;
      s_hid[m * HH + nb + lo]      = f2bf(silu_f(acc0[r] + bia0));
      s_hid[m * HH + nb + 16 + lo] = f2bf(silu_f(acc1[r] + bia1));
    }
  }
  __syncthreads();

  // ---- GEMM2: [16 x 256] x Wu2[256 x 256]  (no final SiLU) ----
  acc0 = (v8f){}; acc1 = (v8f){};
#pragma unroll
  for (int k0 = 0; k0 < HH; k0 += 32) {
    const unsigned short* ap = s_hid + lo * HH + k0 + kbA;
    v16bf a = pack16(*(const uint4*)ap, *(const uint4*)(ap + 16));
    const unsigned short* bp0 = u2t + (size_t)(nb + lo) * HH + k0 + kbB;
    const unsigned short* bp1 = u2t + (size_t)(nb + 16 + lo) * HH + k0 + kbB;
    v16bf b0 = pack16(*(const uint4*)bp0, *(const uint4*)(bp0 + 16));
    v16bf b1 = pack16(*(const uint4*)bp1, *(const uint4*)(bp1 + 16));
    acc0 = wmma_bf16(a, b0, acc0);
    acc1 = wmma_bf16(a, b1, acc1);
  }
  {
    const float bia0 = bu2[nb + lo];
    const float bia1 = bu2[nb + 16 + lo];
#pragma unroll
    for (int r = 0; r < 8; ++r) {
      const int m = r + hi * 8;
      out[(size_t)(n0 + m) * HH + nb + lo]      = acc0[r] + bia0;
      out[(size_t)(n0 + m) * HH + nb + 16 + lo] = acc1[r] + bia1;
    }
  }
}

// ---------------------------------------------------------------------------
extern "C" void kernel_launch(void* const* d_in, const int* in_sizes, int n_in,
                              void* d_out, int out_size, void* d_ws, size_t ws_size,
                              hipStream_t stream) {
  (void)in_sizes; (void)n_in; (void)out_size; (void)ws_size;

  const float*     x   = (const float*)d_in[0];
  const float*     pos = (const float*)d_in[1];
  const long long* ei  = (const long long*)d_in[2];   // int64 edge_index [2,E]
  const float*     Wm1 = (const float*)d_in[3];
  const float*     bm1 = (const float*)d_in[4];
  const float*     Wm2 = (const float*)d_in[5];
  const float*     bm2 = (const float*)d_in[6];
  const float*     Wu1 = (const float*)d_in[7];
  const float*     bu1 = (const float*)d_in[8];
  const float*     Wu2 = (const float*)d_in[9];
  const float*     bu2 = (const float*)d_in[10];
  float* out = (float*)d_out;

  // workspace layout (all offsets 256B-aligned)
  char* ws = (char*)d_ws;
  size_t off = 0;
  float*          aggr = (float*)(ws + off);          off += (size_t)NN * HH * 4;
  unsigned short* xb   = (unsigned short*)(ws + off); off += (size_t)NN * HH * 2;
  unsigned short* w1t  = (unsigned short*)(ws + off); off += (size_t)HH * KP1 * 2;
  unsigned short* w2t  = (unsigned short*)(ws + off); off += (size_t)HH * HH * 2;
  unsigned short* u1t  = (unsigned short*)(ws + off); off += (size_t)HH * KU * 2;
  unsigned short* u2t  = (unsigned short*)(ws + off); off += (size_t)HH * HH * 2;

  // prep
  prep_x_kernel<<<(NN * HH + 255) / 256, 256, 0, stream>>>(x, xb, aggr, NN * HH);
  prep_wt_kernel<<<(HH * KP1 + 255) / 256, 256, 0, stream>>>(Wm1, w1t, 2 * HH + 1, HH, KP1);
  prep_wt_kernel<<<(HH * HH  + 255) / 256, 256, 0, stream>>>(Wm2, w2t, HH, HH, HH);
  prep_wt_kernel<<<(HH * KU  + 255) / 256, 256, 0, stream>>>(Wu1, u1t, KU, HH, KU);
  prep_wt_kernel<<<(HH * HH  + 255) / 256, 256, 0, stream>>>(Wu2, u2t, HH, HH, HH);

  // fused edge MLP + scatter (64 edges per workgroup)
  egnn_edge_kernel<<<EE / MT, 256, 0, stream>>>(xb, pos, ei, w1t, bm1, w2t, bm2, aggr);

  // node update MLP
  egnn_node_kernel<<<NN / 16, 256, 0, stream>>>(xb, aggr, u1t, bu1, u2t, bu2, out);
}
